// PoseWarperNeck_86045374808578
// MI455X (gfx1250) — matrix operand order repaired and checked by reading, not compiled, not run on hardware
//
#include <hip/hip_runtime.h>

typedef __attribute__((ext_vector_type(16))) _Float16 v16h;
typedef __attribute__((ext_vector_type(8)))  _Float16 v8h;
typedef __attribute__((ext_vector_type(8)))  float    v8f;

#define BN_INV 0.99999500003749969f   /* 1/sqrt(1+1e-5) */

// fixed problem geometry (PoseWarperNeck reference)
constexpr int CH  = 96;
constexpr int CW  = 72;
constexpr int CHW = CH * CW;          // 6912, divisible by 128
constexpr int CN  = 20;
constexpr int CB  = 4;
constexpr int CT  = 5;
constexpr int NPIX = CN * CHW;        // 138240 = 1080 * 128

// ---------------------------------------------------------------------------
// transition: 1x1 conv 48->17 + bias, f32 NCHW output
// ---------------------------------------------------------------------------
__global__ void k_transition(const float* __restrict__ x, const float* __restrict__ Wt,
                             const float* __restrict__ bt, float* __restrict__ tr)
{
  int idx = blockIdx.x * blockDim.x + threadIdx.x;
  if (idx >= NPIX) return;
  int n = idx / CHW, p = idx % CHW;
  float xv[48];
#pragma unroll
  for (int ci = 0; ci < 48; ++ci) xv[ci] = x[(n * 48 + ci) * CHW + p];
  for (int c = 0; c < 17; ++c) {
    float a = bt[c];
#pragma unroll
    for (int ci = 0; ci < 48; ++ci) a += xv[ci] * Wt[c * 48 + ci];
    tr[(n * 17 + c) * CHW + p] = a;
  }
}

// ---------------------------------------------------------------------------
// diff = tile(tr[:B]) - tr  -> f16 NHWC, channels padded 17 -> 32
// ---------------------------------------------------------------------------
__global__ void k_diff(const float* __restrict__ tr, _Float16* __restrict__ dif)
{
  int idx = blockIdx.x * blockDim.x + threadIdx.x;
  if (idx >= NPIX) return;
  int n = idx / CHW, p = idx % CHW;
  int b = n % CB;
  _Float16* d = dif + idx * 32;
#pragma unroll
  for (int c = 0; c < 32; ++c) {
    float v = 0.f;
    if (c < 17) v = tr[(b * 17 + c) * CHW + p] - tr[(n * 17 + c) * CHW + p];
    d[c] = (_Float16)v;
  }
}

// ---------------------------------------------------------------------------
// weight pack: f32 OIHW [Cout,Cin,kh,kw] -> f16 [taps][Cin_pad/32][Cout_pad][32]
// ---------------------------------------------------------------------------
__global__ void k_pack_w(const float* __restrict__ src, _Float16* __restrict__ dst,
                         int Cout, int Cin, int taps, int Cin_pad, int Cout_pad)
{
  int idx = blockIdx.x * blockDim.x + threadIdx.x;
  int total = taps * Cin_pad * Cout_pad;
  if (idx >= total) return;
  int co = idx % Cout_pad;
  int t  = idx / Cout_pad;
  int ci = t % Cin_pad;
  int kt = t / Cin_pad;
  float v = 0.f;
  if (ci < Cin && co < Cout) v = src[(co * Cin + ci) * taps + kt];
  int d = ((kt * (Cin_pad >> 5) + (ci >> 5)) * Cout_pad + co) * 32 + (ci & 31);
  dst[d] = (_Float16)v;
}

__global__ void k_zero(float* __restrict__ p, int n)
{
  int i = blockIdx.x * blockDim.x + threadIdx.x;
  if (i < n) p[i] = 0.f;
}

// ---------------------------------------------------------------------------
// WMMA implicit-GEMM conv, specialized at compile time.
//  in   : f16 NHWC [NPIX, CIN_PAD]
//  wB   : f16 [TAPS][CIN_PAD/32][Cout_pad][32]
//  MODE : 0 = BN+ReLU f16 | 1 = BN+resid+ReLU f16 | 2 = plain f32 NCHW | 3 = BN f16
//  block: 256 thr = 8 waves; tile 128 rows x 128 cols; wave = 16-row strip
// ---------------------------------------------------------------------------
template <int TAPS, int CIN_PAD, int MODE>
__global__ void __launch_bounds__(256)
k_conv_wmma(const _Float16* __restrict__ in, const _Float16* __restrict__ wB,
            const float* __restrict__ gamma, const float* __restrict__ beta,
            const _Float16* __restrict__ resid,
            _Float16* __restrict__ outh, float* __restrict__ outf,
            int Cout, int pad, int dil)
{
  constexpr int NCHUNK = CIN_PAD >> 5;
  const int lane = threadIdx.x & 31;
  const int wave = threadIdx.x >> 5;
  const int kg   = lane >> 4;        // K-half group: 0 or 1
  const int l16  = lane & 15;

  const int tile0 = blockIdx.x * 128;        // never crosses an image (6912%128==0)
  const int n  = tile0 / CHW;
  const int p0 = tile0 - n * CHW;
  const int hw = p0 + wave * 16 + l16;
  const int h = hw / CW, w = hw % CW;
  const int coBase = blockIdx.y * 128;

  const v8h zv = {0, 0, 0, 0, 0, 0, 0, 0};
  v8f acc[8];
#pragma unroll
  for (int i = 0; i < 8; ++i) acc[i] = (v8f){0.f, 0.f, 0.f, 0.f, 0.f, 0.f, 0.f, 0.f};

  const int Cout_pad = gridDim.y * 128;

  for (int kt = 0; kt < TAPS; ++kt) {
    int hh = h, ww = w;
    if (TAPS == 9) { hh = h - pad + (kt / 3) * dil; ww = w - pad + (kt % 3) * dil; }
    bool ok = (hh >= 0) & (hh < CH) & (ww >= 0) & (ww < CW);
    int hc = ok ? hh : 0, wc = ok ? ww : 0;
    const _Float16* src = in + (n * CHW + hc * CW + wc) * CIN_PAD;
    const _Float16* wk  = wB + kt * NCHUNK * Cout_pad * 32;

#pragma unroll
    for (int c32 = 0; c32 < NCHUNK; ++c32) {
      const int kbA = c32 * 32 + kg * 8;
      v8h alo = *(const v8h*)(src + kbA);
      v8h ahi = *(const v8h*)(src + kbA + 16);
      alo = ok ? alo : zv;
      ahi = ok ? ahi : zv;
      v16h a;
#pragma unroll
      for (int i = 0; i < 8; ++i) { a[i] = alo[i]; a[8 + i] = ahi[i]; }

      const _Float16* wc32 = wk + c32 * Cout_pad * 32 + kg * 8;
      __builtin_prefetch(wc32 + Cout_pad * 32, 0, 1);   // next weight chunk
#pragma unroll
      for (int nt = 0; nt < 8; ++nt) {
        const _Float16* bp = wc32 + (coBase + nt * 16 + l16) * 32;
        v8h blo = *(const v8h*)(bp);
        v8h bhi = *(const v8h*)(bp + 16);
        v16h b;
#pragma unroll
        for (int i = 0; i < 8; ++i) { b[i] = blo[i]; b[8 + i] = bhi[i]; }
        acc[nt] = __builtin_amdgcn_wmma_f32_16x16x32_f16(
            false, a, false, b, (short)0, acc[nt], false, false);
      }
    }
  }

  // --------------------------- epilogue ---------------------------
  // C/D layout: VGPR g holds row (g + kg*8) of the 16x16 tile, col = l16.
  const int prow = p0 + wave * 16 + kg * 8;  // in-image pixel of VGPR 0
#pragma unroll
  for (int nt = 0; nt < 8; ++nt) {
    const int co = coBase + nt * 16 + l16;
    float sc = 1.f, bs = 0.f;
    if (MODE != 2) { sc = gamma[co] * BN_INV; bs = beta[co]; }
    const bool cok = (MODE != 2) || (co < Cout);
#pragma unroll
    for (int g = 0; g < 8; ++g) {
      float v = acc[nt][g];
      if (MODE == 2) {
        if (cok) outf[(n * Cout + co) * CHW + prow + g] = v;
      } else {
        v = v * sc + bs;
        const int row = tile0 - p0 + prow + g;   // global pixel row
        if (MODE == 1) v += (float)resid[row * 128 + co];
        if (MODE != 3) v = v > 0.f ? v : 0.f;
        outh[row * 128 + co] = (_Float16)v;
      }
    }
  }
}

// ---------------------------------------------------------------------------
// deformable conv (DG=17, cpg=1) + frame_weight/5 accumulation into d_out
// one launch per dilation; threads over (b, h, w); loops t, g, k internally
// ---------------------------------------------------------------------------
__global__ void k_deform(const float* __restrict__ tr, const float* __restrict__ off,
                         const float* __restrict__ dW, const float* __restrict__ fw,
                         float* __restrict__ out, int dil)
{
  int idx = blockIdx.x * blockDim.x + threadIdx.x;
  if (idx >= CB * CHW) return;
  int b = idx / CHW, hw = idx % CHW;
  int h = hw / CW, w = hw % CW;

  float acc[17];
#pragma unroll
  for (int o = 0; o < 17; ++o) acc[o] = 0.f;

  for (int t = 0; t < CT; ++t) {
    int n = t * CB + b;
    float coeff = fw[t] * 0.2f;
    const float* offn = off + n * 306 * CHW + hw;
    const float* trn  = tr + n * 17 * CHW;
    float tmp[17];
#pragma unroll
    for (int o = 0; o < 17; ++o) tmp[o] = 0.f;

    for (int g = 0; g < 17; ++g) {
      const float* img = trn + g * CHW;
      for (int k = 0; k < 9; ++k) {
        int c = (g * 9 + k) * 2;
        float dy = offn[c * CHW];
        float dx = offn[(c + 1) * CHW];
        float py = dy + (float)((k / 3 - 1) * dil + h);
        float px = dx + (float)((k % 3 - 1) * dil + w);
        float y0f = floorf(py), x0f = floorf(px);
        float wy = py - y0f, wx = px - x0f;
        int y0 = (int)y0f, x0 = (int)x0f;
        float s = 0.f;
#pragma unroll
        for (int dyc = 0; dyc < 2; ++dyc) {
#pragma unroll
          for (int dxc = 0; dxc < 2; ++dxc) {
            int yy = y0 + dyc, xx = x0 + dxc;
            bool valid = (yy >= 0) & (yy < CH) & (xx >= 0) & (xx < CW);
            int yc = yy < 0 ? 0 : (yy > CH - 1 ? CH - 1 : yy);
            int xc = xx < 0 ? 0 : (xx > CW - 1 ? CW - 1 : xx);
            float wgt = (dyc ? wy : 1.f - wy) * (dxc ? wx : 1.f - wx);
            float val = img[yc * CW + xc];
            s += valid ? wgt * val : 0.f;
          }
        }
        const float* wrow = dW + g * 9 + k;
#pragma unroll
        for (int o = 0; o < 17; ++o) tmp[o] += s * wrow[o * 153];
      }
    }
#pragma unroll
    for (int o = 0; o < 17; ++o) acc[o] += coeff * tmp[o];
  }
#pragma unroll
  for (int o = 0; o < 17; ++o) out[(b * 17 + o) * CHW + hw] += acc[o];
}

// ---------------------------------------------------------------------------
extern "C" void kernel_launch(void* const* d_in, const int* in_sizes, int n_in,
                              void* d_out, int out_size, void* d_ws, size_t ws_size,
                              hipStream_t stream)
{
  (void)in_sizes; (void)n_in; (void)ws_size;
  const int DIL[5] = {3, 6, 12, 18, 24};

  const float* x       = (const float*)d_in[0];
  const float* fw      = (const float*)d_in[1];
  const float* trans_W = (const float*)d_in[2];
  const float* trans_b = (const float*)d_in[3];
  const float* ds_W    = (const float*)d_in[4];
  const float* ds_g    = (const float*)d_in[5];
  const float* ds_b    = (const float*)d_in[6];
  const float* b0_W1   = (const float*)d_in[7];
  const float* b0_g1   = (const float*)d_in[8];
  const float* b0_b1   = (const float*)d_in[9];
  const float* b0_W2   = (const float*)d_in[10];
  const float* b0_g2   = (const float*)d_in[11];
  const float* b0_b2   = (const float*)d_in[12];
  const float* blk_W1  = (const float*)d_in[13];
  const float* blk_g1  = (const float*)d_in[14];
  const float* blk_b1  = (const float*)d_in[15];
  const float* blk_W2  = (const float*)d_in[16];
  const float* blk_g2  = (const float*)d_in[17];
  const float* blk_b2  = (const float*)d_in[18];
  const float* off_W   = (const float*)d_in[19];
  const float* dcn_W   = (const float*)d_in[20];

  // ---- workspace carve ----
  char* p = (char*)d_ws;
  auto alloc = [&](size_t bytes) -> void* {
    void* r = (void*)p;
    p += (bytes + 255) & ~(size_t)255;
    return r;
  };
  float*    tr    = (float*)   alloc((size_t)CN * 17 * CHW * 4);
  _Float16* dif   = (_Float16*)alloc((size_t)NPIX * 32 * 2);
  _Float16* actA  = (_Float16*)alloc((size_t)NPIX * 128 * 2);
  _Float16* actB  = (_Float16*)alloc((size_t)NPIX * 128 * 2);
  _Float16* actC  = (_Float16*)alloc((size_t)NPIX * 128 * 2);
  float*    offb  = (float*)   alloc((size_t)CN * 306 * CHW * 4);
  const size_t dsSz  = (size_t)1 * 1 * 128 * 32;      // halves
  const size_t b0aSz = (size_t)9 * 1 * 128 * 32;
  const size_t cvSz  = (size_t)9 * 4 * 128 * 32;      // 128->128 3x3
  const size_t offSz = (size_t)9 * 4 * 384 * 32;      // 128->384(306) 3x3
  _Float16* w_ds   = (_Float16*)alloc(dsSz * 2);
  _Float16* w_b0a  = (_Float16*)alloc(b0aSz * 2);
  _Float16* w_main = (_Float16*)alloc(39 * cvSz * 2); // [0]=b0_W2, [1+2i]=W1_i, [2+2i]=W2_i
  _Float16* w_off  = (_Float16*)alloc(5 * offSz * 2);

  dim3 blk(256);
  auto packw = [&](const float* s, _Float16* d, int Co, int Ci, int taps, int Cip, int Cop) {
    int tot = taps * Cip * Cop;
    k_pack_w<<<dim3((tot + 255) / 256), blk, 0, stream>>>(s, d, Co, Ci, taps, Cip, Cop);
  };
  const dim3 g1(NPIX / 128, 1), g3(NPIX / 128, 3);

  // ---- zero output accumulator ----
  k_zero<<<dim3((out_size + 255) / 256), blk, 0, stream>>>((float*)d_out, out_size);

  // ---- transition + diff ----
  k_transition<<<dim3((NPIX + 255) / 256), blk, 0, stream>>>(x, trans_W, trans_b, tr);
  k_diff<<<dim3((NPIX + 255) / 256), blk, 0, stream>>>(tr, dif);

  // ---- pack all weights ----
  packw(ds_W, w_ds, 128, 17, 1, 32, 128);
  packw(b0_W1, w_b0a, 128, 17, 9, 32, 128);
  packw(b0_W2, w_main, 128, 128, 9, 128, 128);
  for (int i = 0; i < 19; ++i) {
    packw(blk_W1 + (size_t)i * 128 * 128 * 9, w_main + (1 + 2 * i) * cvSz, 128, 128, 9, 128, 128);
    packw(blk_W2 + (size_t)i * 128 * 128 * 9, w_main + (2 + 2 * i) * cvSz, 128, 128, 9, 128, 128);
  }
  for (int j = 0; j < 5; ++j)
    packw(off_W + (size_t)j * 306 * 128 * 9, w_off + (size_t)j * offSz, 306, 128, 9, 128, 384);

  // ---- block 0: identity = BN(1x1 conv(diff)); f = basic_block ----
  k_conv_wmma<1, 32, 3><<<g1, blk, 0, stream>>>(
      dif, w_ds, ds_g, ds_b, nullptr, actC, nullptr, 128, 0, 1);
  k_conv_wmma<9, 32, 0><<<g1, blk, 0, stream>>>(
      dif, w_b0a, b0_g1, b0_b1, nullptr, actA, nullptr, 128, 1, 1);
  k_conv_wmma<9, 128, 1><<<g1, blk, 0, stream>>>(
      actA, w_main, b0_g2, b0_b2, actC, actB, nullptr, 128, 1, 1);

  // ---- 19 residual blocks (in-place residual add is one-to-one safe) ----
  for (int i = 0; i < 19; ++i) {
    k_conv_wmma<9, 128, 0><<<g1, blk, 0, stream>>>(
        actB, w_main + (1 + 2 * i) * cvSz, blk_g1 + i * 128, blk_b1 + i * 128,
        nullptr, actA, nullptr, 128, 1, 1);
    k_conv_wmma<9, 128, 1><<<g1, blk, 0, stream>>>(
        actA, w_main + (2 + 2 * i) * cvSz, blk_g2 + i * 128, blk_b2 + i * 128,
        actB, actB, nullptr, 128, 1, 1);
  }

  // ---- per-dilation: offset conv (f32 NCHW) then deform+reduce into out ----
  for (int j = 0; j < 5; ++j) {
    k_conv_wmma<9, 128, 2><<<g3, blk, 0, stream>>>(
        actB, w_off + (size_t)j * offSz, nullptr, nullptr, nullptr,
        nullptr, offb, 306, DIL[j], DIL[j]);
    k_deform<<<dim3((CB * CHW + 255) / 256), blk, 0, stream>>>(
        tr, offb, dcn_W + (size_t)j * 17 * 17 * 9, fw, (float*)d_out, DIL[j]);
  }
}
